// GNN_40836549050953
// MI455X (gfx1250) — matrix operand order repaired and checked by reading, not compiled, not run on hardware
//
#include <hip/hip_runtime.h>
#include <hip/hip_fp16.h>
#include <cstdint>
#include <cstddef>

typedef _Float16 h16;
typedef __attribute__((ext_vector_type(16))) _Float16 v16h;
typedef __attribute__((ext_vector_type(8)))  _Float16 v8h;
typedef __attribute__((ext_vector_type(8)))  float    v8f;

#define VOCAB 30000
#define DIN   300
#define NH    4
#define CHH   300
#define HC    1200
#define HIDD  768
#define TT    800
#define MSLOT (3*TT)        // 2400 active slots (head|rel|tail)
#define NEDGE (2*TT)        // 1600 real edges
#define EAUG  (NEDGE+MSLOT) // + self-loop candidates
#define LEAK  0.2f

// K strides padded to multiples of 64 (double-buffered 64-wide K chunks)
#define KP_D   320   // 300  -> 320
#define KP_HC  1216  // 1200 -> 1216
#define KP_F   960   // 900  -> 960

// LDS staging geometry for the B tile: 32 cols x 64 K halves per chunk
#define LDS_ROWB  144                   // 128 B payload + 16 B bank-spread pad
#define LDS_BUF   (32 * LDS_ROWB)       // 4608 B per buffer

// lds_addr must be derived from the shared array (ptrtoint) so the array is
// captured by the asm and the compiler keeps the subsequent ds_load reads.
// Low 32 bits of a flat LDS address == LDS byte offset (ISA 10.2).
__device__ __forceinline__ void async_load_b128(unsigned lds_addr, const h16* gaddr)
{
    asm volatile("global_load_async_to_lds_b128 %0, %1, off"
                 :: "v"(lds_addr), "v"((unsigned long long)(uintptr_t)gaddr)
                 : "memory");
}
__device__ __forceinline__ void wait_async0()
{
    asm volatile("s_wait_asynccnt 0" ::: "memory");
}

// ---------------------------------------------------------------------------
// WMMA GEMM: C[M,N] = A[M,K] * B[K,N] (+ bias). A fp16 row-major (lda = K
// padded to 64), B pre-transposed as Bt[Npad,K]. Block tile 128x32: 8 waves,
// each wave owns a 16x32 strip (two v_wmma per K-step sharing one A frag).
// B tile staged through LDS with double-buffered global_load_async_to_lds_b128.
// ---------------------------------------------------------------------------
__global__ __launch_bounds__(256)
void wmma_gemm(const h16* __restrict__ A, int lda,
               const h16* __restrict__ Bt, int ldb,
               const float* __restrict__ bias,
               float* __restrict__ Cf, h16* __restrict__ Ch, int ldc,
               int Mrows, int Nreal, int K)
{
    __shared__ __align__(16) unsigned char ldsB[2 * LDS_BUF];

    const int tid  = threadIdx.x;
    const int wave = tid >> 5;
    const int lane = tid & 31;
    const int g    = lane >> 4;        // half-wave group
    const int r    = lane & 15;

    const int n0 = blockIdx.x * 32;
    const int m0 = blockIdx.y * 128 + wave * 16;

    int rowA = m0 + r;
    if (rowA >= Mrows) rowA = Mrows - 1;          // clamp; store is guarded
    const h16* Ap = A + (size_t)rowA * lda + g * 8;

    // cooperative staging: thread i moves 16 B of the 32x64 B chunk
    const int srow = tid >> 3, sseg = tid & 7;
    const h16* Bg = Bt + (size_t)(n0 + srow) * ldb + sseg * 8;
    const unsigned stage =
        (unsigned)(uintptr_t)(ldsB + srow * LDS_ROWB + sseg * 16);

    const int nk = K >> 6;                         // 64-wide K chunks
    async_load_b128(stage, Bg);                    // chunk 0 -> buffer 0

    v8f acc0 = {}, acc1 = {};
    for (int c = 0; c < nk; ++c) {
        wait_async0();
        __syncthreads();                           // B chunk c visible to all waves
        if (c + 1 < nk)
            async_load_b128(stage + ((c + 1) & 1) * LDS_BUF,
                            Bg + (size_t)(c + 1) * 64);

        const int k0 = c * 64;
        const unsigned bufo = (unsigned)((c & 1) * LDS_BUF);
        __builtin_prefetch(Ap + k0 + 64, 0, 3);    // next A chunk
#pragma unroll
        for (int kk = 0; kk < 64; kk += 32) {
            // A 16x32 frag: lane(g,r) = row r, K {k0+kk+8g..} and {k0+kk+16+8g..}
            v8h a0 = *(const v8h*)(Ap + k0 + kk);
            v8h a1 = *(const v8h*)(Ap + k0 + kk + 16);
            v16h a;
#pragma unroll
            for (int t = 0; t < 8; ++t) { a[t] = a0[t]; a[8 + t] = a1[t]; }

            // B 32x16 frags from LDS: lane(g,r) = col r (+16), K {kk+8g..} and {kk+16+8g..}
            const unsigned o0 = bufo + (unsigned)(r * LDS_ROWB + kk * 2 + g * 16);
            const unsigned o1 = o0 + 16 * LDS_ROWB;
            v8h b00 = *(const v8h*)(ldsB + o0);
            v8h b01 = *(const v8h*)(ldsB + o0 + 32);
            v8h b10 = *(const v8h*)(ldsB + o1);
            v8h b11 = *(const v8h*)(ldsB + o1 + 32);
            v16h b0, b1;
#pragma unroll
            for (int t = 0; t < 8; ++t) {
                b0[t] = b00[t]; b0[8 + t] = b01[t];
                b1[t] = b10[t]; b1[8 + t] = b11[t];
            }
            acc0 = __builtin_amdgcn_wmma_f32_16x16x32_f16(
                       false, a, false, b0, (short)0, acc0, false, false);
            acc1 = __builtin_amdgcn_wmma_f32_16x16x32_f16(
                       false, a, false, b1, (short)0, acc1, false, false);
        }
    }

#pragma unroll
    for (int s = 0; s < 2; ++s) {
        const v8f acc = s ? acc1 : acc0;
        const int col = n0 + s * 16 + r;
        float bv = (bias != nullptr && col < Nreal) ? bias[col] : 0.f;
#pragma unroll
        for (int v = 0; v < 8; ++v) {
            int row = m0 + v + 8 * g;              // C/D frag layout
            if (row < Mrows && col < Nreal) {
                float x = acc[v] + bv;
                if (Cf) Cf[(size_t)row * ldc + col] = x;
                if (Ch) Ch[(size_t)row * ldc + col] = (h16)x;
            }
        }
    }
}

// --------------------------- helper kernels --------------------------------

// dst[Npad,Kp] fp16 = transpose(src[K,N]) zero-padded
__global__ void transpose_h16(const float* __restrict__ src, h16* __restrict__ dst,
                              int K, int N, int Kp, int Npad)
{
    int idx = blockIdx.x * blockDim.x + threadIdx.x;
    int total = Npad * Kp;
    if (idx >= total) return;
    int n = idx / Kp, k = idx - n * Kp;
    float v = (n < N && k < K) ? src[(size_t)k * N + n] : 0.f;
    dst[idx] = (h16)v;
}

__global__ void zero_h16(h16* p, int n)
{
    int i = blockIdx.x * blockDim.x + threadIdx.x;
    if (i < n) p[i] = (h16)0.f;
}

__global__ void init_rep(int* repSlot)
{
    int i = blockIdx.x * blockDim.x + threadIdx.x;
    if (i < VOCAB) repSlot[i] = 0x7FFFFFFF;
}

// slots: [0,T)=head, [T,2T)=rel, [2T,3T)=tail; edge e: src slot e, dst slot T+e
__global__ void build_nodes(const int* __restrict__ kg, int* __restrict__ nodes,
                            int* __restrict__ repSlot)
{
    int i = blockIdx.x * blockDim.x + threadIdx.x;
    if (i >= MSLOT) return;
    int v;
    if (i < TT)            v = kg[i * 3 + 0];
    else if (i < 2 * TT)   v = kg[(i - TT) * 3 + 1];
    else                   v = kg[(i - 2 * TT) * 3 + 2];
    nodes[i] = v;
    atomicMin(&repSlot[v], i);
}

__global__ void gather_embed(const float* __restrict__ emb, const int* __restrict__ nodes,
                             h16* __restrict__ Ax)
{
    int i = blockIdx.x;
    int node = nodes[i];
    for (int k = threadIdx.x; k < KP_D; k += blockDim.x)
        Ax[(size_t)i * KP_D + k] = (k < DIN) ? (h16)emb[(size_t)node * DIN + k] : (h16)0.f;
}

__global__ void init_attn(unsigned* __restrict__ ml, float* __restrict__ dn,
                          float* __restrict__ hacc)
{
    size_t i = (size_t)blockIdx.x * blockDim.x + threadIdx.x;
    if (i < (size_t)VOCAB * NH) { ml[i] = 0u; dn[i] = 0.f; }
    if (i < (size_t)MSLOT * HC) hacc[i] = 0.f;
}

__device__ inline unsigned f2ord(float f)
{
    unsigned u = __float_as_uint(f);
    return (u & 0x80000000u) ? ~u : (u | 0x80000000u);
}
__device__ inline float ord2f(unsigned u)
{
    return __uint_as_float((u & 0x80000000u) ? (u & 0x7FFFFFFFu) : ~u);
}

// pass 1: per-(edge,head) logit = att . leaky(xl[s]+xr[d]); segment-max by dst vocab
__global__ void attn_logits(const float* __restrict__ xl, const float* __restrict__ xr,
                            const float* __restrict__ att,
                            const int* __restrict__ nodes, const int* __restrict__ repSlot,
                            float* __restrict__ lg, unsigned* __restrict__ ml)
{
    int idx = blockIdx.x * blockDim.x + threadIdx.x;
    if (idx >= EAUG * NH) return;
    int e = idx / NH, h = idx - e * NH;
    int s, d;
    if (e < NEDGE) { s = e; d = TT + e; }
    else {
        int i = e - NEDGE;
        if (repSlot[nodes[i]] != i) { lg[idx] = 0.f; return; }  // duplicate self-loop
        s = i; d = i;
    }
    const float* pl = xl + (size_t)s * HC + h * CHH;
    const float* pr = xr + (size_t)d * HC + h * CHH;
    const float* pa = att + h * CHH;
    float acc = 0.f;
    for (int c = 0; c < CHH; ++c) {
        float v = pl[c] + pr[c];
        v = v > 0.f ? v : LEAK * v;
        acc += pa[c] * v;
    }
    lg[idx] = acc;
    atomicMax(&ml[(size_t)nodes[d] * NH + h], f2ord(acc));
}

// pass 2: ex = exp(logit - max); segment-sum by dst vocab
__global__ void attn_exp(const int* __restrict__ nodes, const int* __restrict__ repSlot,
                         float* __restrict__ lg, const unsigned* __restrict__ ml,
                         float* __restrict__ dn)
{
    int idx = blockIdx.x * blockDim.x + threadIdx.x;
    if (idx >= EAUG * NH) return;
    int e = idx / NH, h = idx - e * NH;
    int d;
    if (e < NEDGE) d = TT + e;
    else { int i = e - NEDGE; if (repSlot[nodes[i]] != i) return; d = i; }
    int dv = nodes[d];
    float ex = __expf(lg[idx] - ord2f(ml[(size_t)dv * NH + h]));
    lg[idx] = ex;
    atomicAdd(&dn[(size_t)dv * NH + h], ex);
}

// pass 3: hacc[rep(dst)] += alpha * xl[src]
__global__ void attn_aggregate(const int* __restrict__ nodes, const int* __restrict__ repSlot,
                               const float* __restrict__ lg, const float* __restrict__ dn,
                               const float* __restrict__ xl, float* __restrict__ hacc)
{
    int e = blockIdx.x;
    int s, d;
    if (e < NEDGE) { s = e; d = TT + e; }
    else { int i = e - NEDGE; if (repSlot[nodes[i]] != i) return; s = i; d = i; }
    int dv = nodes[d];
    int rr = repSlot[dv];
    __shared__ float alpha[NH];
    if (threadIdx.x < NH)
        alpha[threadIdx.x] = lg[(size_t)e * NH + threadIdx.x]
                           / dn[(size_t)dv * NH + threadIdx.x];
    __syncthreads();
    for (int j = threadIdx.x; j < HC; j += blockDim.x)
        atomicAdd(&hacc[(size_t)rr * HC + j], alpha[j / CHH] * xl[(size_t)s * HC + j]);
}

// broadcast rep row + bias + relu -> fp16 activations (K-padded to KP_HC)
__global__ void attn_finalize(const int* __restrict__ nodes, const int* __restrict__ repSlot,
                              const float* __restrict__ hacc, const float* __restrict__ bias,
                              h16* __restrict__ hh)
{
    int i = blockIdx.x;
    int rr = repSlot[nodes[i]];
    for (int j = threadIdx.x; j < KP_HC; j += blockDim.x) {
        float v = 0.f;
        if (j < HC) v = fmaxf(hacc[(size_t)rr * HC + j] + bias[j], 0.f);
        hh[(size_t)i * KP_HC + j] = (h16)v;
    }
}

// F[t] = [w[head_t] | w[rel_t] | w[tail_t]]  (rows t, T+t, 2T+t of wh)
__global__ void build_F(const h16* __restrict__ wh, h16* __restrict__ F)
{
    int t = blockIdx.x;
    for (int j = threadIdx.x; j < KP_F; j += blockDim.x) {
        h16 v = (h16)0.f;
        if (j < 3 * DIN) {
            int part = j / DIN, c = j - part * DIN;
            v = wh[(size_t)(part * TT + t) * KP_D + c];
        }
        F[(size_t)t * KP_F + j] = v;
    }
}

// ---------------------------------------------------------------------------
extern "C" void kernel_launch(void* const* d_in, const int* in_sizes, int n_in,
                              void* d_out, int out_size, void* d_ws, size_t ws_size,
                              hipStream_t stream)
{
    (void)in_sizes; (void)n_in; (void)out_size; (void)ws_size;
    const int*   kg    = (const int*)d_in[0];
    const float* emb   = (const float*)d_in[1];
    const float* Wl1   = (const float*)d_in[2];
    const float* bl1   = (const float*)d_in[3];
    const float* Wr1   = (const float*)d_in[4];
    const float* br1   = (const float*)d_in[5];
    const float* att1  = (const float*)d_in[6];
    const float* bias1 = (const float*)d_in[7];
    const float* Wl2   = (const float*)d_in[8];
    const float* bl2   = (const float*)d_in[9];
    const float* Wr2   = (const float*)d_in[10];
    const float* br2   = (const float*)d_in[11];
    const float* att2  = (const float*)d_in[12];
    const float* bias2 = (const float*)d_in[13];
    const float* Wp1   = (const float*)d_in[14];
    const float* bp1   = (const float*)d_in[15];
    const float* Wp2   = (const float*)d_in[16];
    const float* bp2   = (const float*)d_in[17];
    const float* Wlii  = (const float*)d_in[18];
    float* out = (float*)d_out;

    // ---- workspace carve-up (~60 MB) ----
    char* base = (char*)d_ws;
    size_t off = 0;
    auto alloc = [&](size_t bytes) {
        void* p = base + off;
        off = (off + bytes + 255) & ~(size_t)255;
        return p;
    };
    h16* WlT1  = (h16*)alloc((size_t)1216 * KP_D  * 2);
    h16* WrT1  = (h16*)alloc((size_t)1216 * KP_D  * 2);
    h16* WlT2  = (h16*)alloc((size_t)1216 * KP_HC * 2);
    h16* WrT2  = (h16*)alloc((size_t)1216 * KP_HC * 2);
    h16* WpT1  = (h16*)alloc((size_t)320  * KP_HC * 2);   // Npad 320 (Nreal 300)
    h16* WpT2  = (h16*)alloc((size_t)320  * KP_D  * 2);
    h16* WliiT = (h16*)alloc((size_t)HIDD * KP_F  * 2);
    h16* Ax    = (h16*)alloc((size_t)MSLOT * KP_D  * 2);
    h16* hh    = (h16*)alloc((size_t)MSLOT * KP_HC * 2);
    h16* w1h   = (h16*)alloc((size_t)MSLOT * KP_D  * 2);
    h16* wh    = (h16*)alloc((size_t)MSLOT * KP_D  * 2);
    h16* Fm    = (h16*)alloc((size_t)TT * KP_F * 2);
    float* xl   = (float*)alloc((size_t)MSLOT * HC * 4);
    float* xr   = (float*)alloc((size_t)MSLOT * HC * 4);
    float* hacc = (float*)alloc((size_t)MSLOT * HC * 4);
    float* dn   = (float*)alloc((size_t)VOCAB * NH * 4);
    float* lg   = (float*)alloc((size_t)EAUG * NH * 4);
    unsigned* ml = (unsigned*)alloc((size_t)VOCAB * NH * 4);
    int* repSlot = (int*)alloc((size_t)VOCAB * 4);
    int* nodes   = (int*)alloc((size_t)MSLOT * 4);

    auto gemm = [&](const h16* A, int lda, const h16* Bt, int ldb, const float* bias,
                    float* Cf, h16* Ch, int ldc, int Mr, int Npad, int Nreal, int K) {
        dim3 grid(Npad / 32, (Mr + 127) / 128);
        wmma_gemm<<<grid, 256, 0, stream>>>(A, lda, Bt, ldb, bias, Cf, Ch, ldc, Mr, Nreal, K);
    };
    auto xpose = [&](const float* src, h16* dst, int K, int N, int Kp, int Npad) {
        int total = Npad * Kp;
        transpose_h16<<<(total + 255) / 256, 256, 0, stream>>>(src, dst, K, N, Kp, Npad);
    };

    // ---- one-time (per launch) weight transpose + fp16 convert ----
    xpose(Wl1, WlT1, DIN, HC, KP_D, 1216);
    xpose(Wr1, WrT1, DIN, HC, KP_D, 1216);
    xpose(Wl2, WlT2, HC, HC, KP_HC, 1216);
    xpose(Wr2, WrT2, HC, HC, KP_HC, 1216);
    xpose(Wp1, WpT1, HC, DIN, KP_HC, 320);
    xpose(Wp2, WpT2, DIN, DIN, KP_D, 320);
    xpose(Wlii, WliiT, 3 * DIN, HIDD, KP_F, HIDD);

    const int attn_grid = (EAUG * NH + 127) / 128;
    const int init_grid = ((size_t)MSLOT * HC + 255) / 256;

    for (int b = 0; b < 2; ++b) {
        const int* kgb = kg + (size_t)b * TT * 3;
        float* outb = out + (size_t)b * TT * HIDD;

        // active slot list + representative slot per vocab id
        init_rep<<<(VOCAB + 255) / 256, 256, 0, stream>>>(repSlot);
        build_nodes<<<(MSLOT + 255) / 256, 256, 0, stream>>>(kgb, nodes, repSlot);
        gather_embed<<<MSLOT, 128, 0, stream>>>(emb, nodes, Ax);

        // ---- layer 1 ----
        gemm(Ax, KP_D, WlT1, KP_D, bl1, xl, nullptr, HC, MSLOT, 1216, HC, KP_D);
        gemm(Ax, KP_D, WrT1, KP_D, br1, xr, nullptr, HC, MSLOT, 1216, HC, KP_D);
        init_attn<<<init_grid, 256, 0, stream>>>(ml, dn, hacc);
        attn_logits<<<attn_grid, 128, 0, stream>>>(xl, xr, att1, nodes, repSlot, lg, ml);
        attn_exp<<<attn_grid, 128, 0, stream>>>(nodes, repSlot, lg, ml, dn);
        attn_aggregate<<<EAUG, 256, 0, stream>>>(nodes, repSlot, lg, dn, xl, hacc);
        attn_finalize<<<MSLOT, 256, 0, stream>>>(nodes, repSlot, hacc, bias1, hh);

        // ---- layer 2 ----
        gemm(hh, KP_HC, WlT2, KP_HC, bl2, xl, nullptr, HC, MSLOT, 1216, HC, KP_HC);
        gemm(hh, KP_HC, WrT2, KP_HC, br2, xr, nullptr, HC, MSLOT, 1216, HC, KP_HC);
        init_attn<<<init_grid, 256, 0, stream>>>(ml, dn, hacc);
        attn_logits<<<attn_grid, 128, 0, stream>>>(xl, xr, att2, nodes, repSlot, lg, ml);
        attn_exp<<<attn_grid, 128, 0, stream>>>(nodes, repSlot, lg, ml, dn);
        attn_aggregate<<<EAUG, 256, 0, stream>>>(nodes, repSlot, lg, dn, xl, hacc);
        attn_finalize<<<MSLOT, 256, 0, stream>>>(nodes, repSlot, hacc, bias2, hh);

        // ---- projections ----
        zero_h16<<<((MSLOT * KP_D) + 255) / 256, 256, 0, stream>>>(w1h, MSLOT * KP_D);
        gemm(hh, KP_HC, WpT1, KP_HC, bp1, nullptr, w1h, KP_D, MSLOT, 320, DIN, KP_HC);
        zero_h16<<<((MSLOT * KP_D) + 255) / 256, 256, 0, stream>>>(wh, MSLOT * KP_D);
        gemm(w1h, KP_D, WpT2, KP_D, bp2, nullptr, wh, KP_D, MSLOT, 320, DIN, KP_D);

        // ---- final: concat(w[head],w[rel],w[tail]) @ Wlii ----
        build_F<<<TT, 256, 0, stream>>>(wh, Fm);
        gemm(Fm, KP_F, WliiT, KP_F, nullptr, outb, nullptr, HIDD, TT, HIDD, HIDD, KP_F);
    }
}
